// Decoder_33646773797599
// MI455X (gfx1250) — matrix operand order repaired and checked by reading, not compiled, not run on hardware
//
#include <hip/hip_runtime.h>
#include <math.h>

#define LSLOPE 0.01f

typedef __bf16 bf16;
typedef __attribute__((ext_vector_type(16))) __bf16 v16bf;
typedef __attribute__((ext_vector_type(8)))  float  v8f;
typedef __attribute__((ext_vector_type(8)))  unsigned int v8u;

__device__ __forceinline__ float leaky(float x) { return x >= 0.f ? x : LSLOPE * x; }

// ---------------------------------------------------------------------------
// Pack W [64 (k), 64 (n)] f32 into bf16 WMMA B-operand fragments:
// packB[((ks*4 + t)*32 + lane)*16 + 2v + {0,1}] = W[(ks*32 + 2v + 8*hi + 8*(v>>2)) + {0,1}][16t + lo]
// 256 threads, one fragment (16 bf16 = 32 B) per thread.
// ---------------------------------------------------------------------------
__global__ void pack_B_kernel(const float* __restrict__ W, bf16* __restrict__ packB)
{
    const int tid  = threadIdx.x;        // 0..255
    const int lane = tid & 31;
    const int frag = tid >> 5;           // ks*4 + t
    const int ks   = frag >> 2;
    const int t    = frag & 3;
    const int lo   = lane & 15;
    const int hi   = lane >> 4;
    const int c    = 16 * t + lo;
    bf16* dst = packB + (size_t)tid * 16;
#pragma unroll
    for (int v = 0; v < 8; ++v) {
        const int k0 = ks * 32 + 2 * v + 8 * hi + 8 * (v >> 2);
        dst[2 * v]     = (bf16)W[(size_t)k0 * 64 + c];
        dst[2 * v + 1] = (bf16)W[(size_t)(k0 + 1) * 64 + c];
    }
}

// ---------------------------------------------------------------------------
// K1: ea = leaky(gather(edge_z, mask) @ W + b), fused per-row mean.
// One wave per 16-row M tile; N=64 (4 tiles), K=64 (2 k-steps of 32).
// ---------------------------------------------------------------------------
__global__ void edge_gemm_wmma(const float* __restrict__ Efeat,
                               const int* __restrict__ mask,
                               const bf16* __restrict__ packB,
                               const float* __restrict__ bias,
                               float* __restrict__ ea,
                               float* __restrict__ rowmean,
                               int Eb)
{
    const int lane    = threadIdx.x & 31;
    const int wave    = (blockIdx.x * blockDim.x + threadIdx.x) >> 5;
    const int rowBase = wave * 16;
    if (rowBase >= Eb) return;

    const int lo = lane & 15;
    const int hi = lane >> 4;

    int r = rowBase + lo;
    if (r >= Eb) r = Eb - 1;
    const int srcRow = mask ? mask[r] : r;
    const float4* arow4 = (const float4*)(Efeat + (size_t)srcRow * 64);
    const v8u* bfrag = (const v8u*)packB;

    v8f acc[4] = {};

#pragma unroll
    for (int ks = 0; ks < 2; ++ks) {
        const float4 p0 = arow4[ks * 8 + 2 * hi + 0];  // K = ks*32 + 8hi + 0..3
        const float4 p1 = arow4[ks * 8 + 2 * hi + 1];  // K = ks*32 + 8hi + 4..7
        const float4 p2 = arow4[ks * 8 + 2 * hi + 4];  // K = ks*32 + 16 + 8hi + 0..3
        const float4 p3 = arow4[ks * 8 + 2 * hi + 5];  // K = ks*32 + 16 + 8hi + 4..7
        v16bf a;
        a[0]  = (bf16)p0.x; a[1]  = (bf16)p0.y; a[2]  = (bf16)p0.z; a[3]  = (bf16)p0.w;
        a[4]  = (bf16)p1.x; a[5]  = (bf16)p1.y; a[6]  = (bf16)p1.z; a[7]  = (bf16)p1.w;
        a[8]  = (bf16)p2.x; a[9]  = (bf16)p2.y; a[10] = (bf16)p2.z; a[11] = (bf16)p2.w;
        a[12] = (bf16)p3.x; a[13] = (bf16)p3.y; a[14] = (bf16)p3.z; a[15] = (bf16)p3.w;
#pragma unroll
        for (int t = 0; t < 4; ++t) {
            const v8u raw = bfrag[(ks * 4 + t) * 32 + lane];
            const v16bf b = __builtin_bit_cast(v16bf, raw);
            acc[t] = __builtin_amdgcn_wmma_f32_16x16x32_bf16(
                false, a, false, b, (short)0, acc[t], false, false);
        }
    }

    float rsum[8] = {0.f, 0.f, 0.f, 0.f, 0.f, 0.f, 0.f, 0.f};
#pragma unroll
    for (int t = 0; t < 4; ++t) {
        const int c    = 16 * t + lo;
        const float bc = bias[c];
#pragma unroll
        for (int v = 0; v < 8; ++v) {
            const int row   = rowBase + v + 8 * hi;
            const float val = leaky(acc[t][v] + bc);
            rsum[v] += val;
            if (row < Eb)
                ea[(size_t)row * 64 + c] = val;
        }
    }

    // Reduce each row-sum across the 16 column lanes (xor masks stay in-group).
#pragma unroll
    for (int v = 0; v < 8; ++v) {
        float s = rsum[v];
        s += __shfl_xor(s, 1);
        s += __shfl_xor(s, 2);
        s += __shfl_xor(s, 4);
        s += __shfl_xor(s, 8);
        const int row = rowBase + v + 8 * hi;
        if (lo == 0 && row < Eb)
            rowmean[row] = s * (1.0f / 64.0f);
    }
}

// m[e] = rowmean[sel[e]]  (adj_sel is identity in practice; kept for fidelity)
__global__ void mean_gather(const float* __restrict__ rowmean, const int* __restrict__ sel,
                            float* __restrict__ m, int Eb)
{
    const int e = blockIdx.x * blockDim.x + threadIdx.x;
    if (e < Eb) m[e] = rowmean[sel ? sel[e] : e];
}

__global__ void fill_kernel(float* __restrict__ p, float v, int n)
{
    const int i = blockIdx.x * blockDim.x + threadIdx.x;
    if (i < n) p[i] = v;
}

// Zero only columns 0..127 of out; the full branch writes 128..191 everywhere.
__global__ void zero_cols(float* __restrict__ out, int Efull)
{
    const size_t idx   = (size_t)blockIdx.x * blockDim.x + threadIdx.x;
    const size_t total = (size_t)Efull * 32;  // 32 float4 per edge
    if (idx >= total) return;
    const size_t e = idx >> 5;
    const int    q = (int)(idx & 31);
    const float4 z = {0.f, 0.f, 0.f, 0.f};
    ((float4*)(out + e * 192))[q] = z;
}

// deg accumulation (each undirected edge contributes m to both endpoints)
__global__ void deg_accum(const int* __restrict__ srcI, const int* __restrict__ dstI,
                          const float* __restrict__ m, float* __restrict__ deg, int Eb)
{
    const int e = blockIdx.x * blockDim.x + threadIdx.x;
    if (e >= Eb) return;
    const float w = m[e];
    atomicAdd(&deg[srcI[e]], w);
    atomicAdd(&deg[dstI[e]], w);
}

// deg -> dinv in place
__global__ void dinv_kernel(float* __restrict__ deg, int n)
{
    const int i = blockIdx.x * blockDim.x + threadIdx.x;
    if (i < n) {
        const float d = deg[i];
        deg[i] = (d > 0.f) ? rsqrtf(d) : 0.f;
    }
}

// h = concat(node[idx], weather) @ W   (x: [n,144], W: [144,outDim])
__global__ void feat_gemm(const float* __restrict__ node, const int* __restrict__ idx,
                          const float* __restrict__ wthr, const float* __restrict__ W,
                          int outDim, float* __restrict__ h, int n)
{
    const int i  = blockIdx.x;
    const int o  = threadIdx.x;  // blockDim.x == outDim
    const int ni = idx ? idx[i] : i;
    const float* nrow = node + (size_t)ni * 128;
    const float* wrow = wthr + (size_t)i * 16;
    float acc = 0.f;
    for (int k = 0; k < 128; ++k) acc += nrow[k] * W[(size_t)k * outDim + o];
    for (int k = 0; k < 16;  ++k) acc += wrow[k] * W[(size_t)(128 + k) * outDim + o];
    h[(size_t)i * outDim + o] = acc;
}

// xh = leaky(M @ h + b); M[d,s] = dinv[d]*m(d,s)*dinv[s], diag dinv^2 (complete graph).
__global__ void gcn_agg(const float* __restrict__ h, const float* __restrict__ m,
                        const float* __restrict__ dinv, const float* __restrict__ bias,
                        float* __restrict__ xh, int n, int outDim)
{
    __shared__ float coef[128];
    const int d = blockIdx.x;
    const int o = threadIdx.x;
    const float dv = dinv[d];
    float acc = 0.f;

    for (int s0 = 0; s0 < n; s0 += blockDim.x) {
        const int s = s0 + o;
        float c = 0.f;
        if (s < n) {
            if (s == d) {
                c = dv * dv;  // self-loop, weight 1
            } else {
                const int i = (d < s) ? d : s;
                const int j = (d < s) ? s : d;
                const size_t e = (size_t)i * (size_t)(2 * n - i - 1) / 2 + (size_t)(j - i - 1);
                c = dv * dinv[s] * m[e];
            }
        }
        coef[o] = c;
        __syncthreads();
        const int lim = (n - s0 < (int)blockDim.x) ? (n - s0) : (int)blockDim.x;
        for (int t = 0; t < lim; ++t)
            acc += coef[t] * h[(size_t)(s0 + t) * outDim + o];
        __syncthreads();
    }
    xh[(size_t)d * outDim + o] = leaky(acc + bias[o]);
}

// out[emask(e), colBase + :64] = (sigmoid(xh[i].xh[j]) * cW + cb + 1) * ea[e]
__global__ void finalize(const float* __restrict__ ea, const float* __restrict__ xh,
                         int outDim, const int* __restrict__ srcI, const int* __restrict__ dstI,
                         const int* __restrict__ emask, const float* __restrict__ cW,
                         const float* __restrict__ cb, float* __restrict__ out,
                         int colBase, int Eb)
{
    const int e = blockIdx.x * blockDim.x + threadIdx.x;
    if (e >= Eb) return;
    const int i = srcI[e];
    const int j = dstI[e];
    const float4* xi = (const float4*)(xh + (size_t)i * outDim);
    const float4* xj = (const float4*)(xh + (size_t)j * outDim);
    float dot = 0.f;
    for (int q = 0; q < (outDim >> 2); ++q) {
        const float4 u = xi[q], w = xj[q];
        dot += u.x * w.x + u.y * w.y + u.z * w.z + u.w * w.w;
    }
    const float t = 1.f / (1.f + expf(-dot));

    const size_t eo = emask ? (size_t)emask[e] : (size_t)e;
    const float4* earow = (const float4*)(ea + (size_t)e * 64);
    const float4* cw4 = (const float4*)cW;
    const float4* cb4 = (const float4*)cb;
    float4* orow = (float4*)(out + eo * 192 + colBase);
#pragma unroll
    for (int q = 0; q < 16; ++q) {
        const float4 v = earow[q];
        const float4 w = cw4[q];
        const float4 B = cb4[q];
        float4 o;
        o.x = (t * w.x + B.x + 1.f) * v.x;
        o.y = (t * w.y + B.y + 1.f) * v.y;
        o.z = (t * w.z + B.z + 1.f) * v.z;
        o.w = (t * w.w + B.w + 1.f) * v.w;
        orow[q] = o;
    }
}

// ---------------------------------------------------------------------------
extern "C" void kernel_launch(void* const* d_in, const int* in_sizes, int n_in,
                              void* d_out, int out_size, void* d_ws, size_t ws_size,
                              hipStream_t stream)
{
    (void)in_sizes; (void)n_in; (void)ws_size; (void)out_size;
    const int N = 1024, N_HUB = 512, N_TOP = 256;
    const int E_FULL = N * (N - 1) / 2;
    const int E_HUB  = N_HUB * (N_HUB - 1) / 2;
    const int E_TOP  = N_TOP * (N_TOP - 1) / 2;

    const float* node   = (const float*)d_in[0];
    const float* wthr   = (const float*)d_in[1];
    const float* hub_w  = (const float*)d_in[2];
    const float* top_w  = (const float*)d_in[3];
    const float* edge_z = (const float*)d_in[4];
    const int* adj      = (const int*)d_in[5];
    const int* hub_adj  = (const int*)d_in[6];
    const int* top_adj  = (const int*)d_in[7];
    const int* hub_idx  = (const int*)d_in[8];
    const int* top_idx  = (const int*)d_in[9];
    const int* sel_full = (const int*)d_in[10];
    const int* sel_hub  = (const int*)d_in[11];
    const int* sel_top  = (const int*)d_in[12];
    const int* hub_mask = (const int*)d_in[13];
    const int* top_mask = (const int*)d_in[14];

    float* out = (float*)d_out;
    {
        const size_t total = (size_t)E_FULL * 32;
        const int blocks = (int)((total + 255) / 256);
        zero_cols<<<blocks, 256, 0, stream>>>(out, E_FULL);
    }

    float* ws = (float*)d_ws;
    size_t off = 0;
    auto alloc = [&](size_t nf) -> float* {
        float* p = ws + off;
        off += (nf + 63) & ~((size_t)63);
        return p;
    };

    struct Branch {
        int n, Eb, outDim, colBase;
        const int *mask, *adjp, *sel, *idx;
        const float *w;
        const float *dgW, *dgb, *gW, *gb, *cW, *cb;
    };
    const Branch br[3] = {
        { N_TOP, E_TOP, 128, 0,   top_mask, top_adj, sel_top, top_idx, top_w,
          (const float*)d_in[15], (const float*)d_in[16], (const float*)d_in[17],
          (const float*)d_in[18], (const float*)d_in[19], (const float*)d_in[20] },
        { N_HUB, E_HUB, 64, 64,   hub_mask, hub_adj, sel_hub, hub_idx, hub_w,
          (const float*)d_in[21], (const float*)d_in[22], (const float*)d_in[23],
          (const float*)d_in[24], (const float*)d_in[25], (const float*)d_in[26] },
        { N, E_FULL, 64, 128,     nullptr, adj, sel_full, nullptr, wthr,
          (const float*)d_in[27], (const float*)d_in[28], (const float*)d_in[29],
          (const float*)d_in[30], (const float*)d_in[31], (const float*)d_in[32] },
    };

    for (int bi = 0; bi < 3; ++bi) {
        const Branch& B = br[bi];
        float* ea      = alloc((size_t)B.Eb * 64);
        float* rowmean = alloc((size_t)B.Eb);
        float* m       = alloc((size_t)B.Eb);
        float* deg     = alloc((size_t)B.n);          // becomes dinv in place
        float* h       = alloc((size_t)B.n * B.outDim);
        float* xh      = alloc((size_t)B.n * B.outDim);
        bf16*  packB   = (bf16*)alloc(2048);          // 8 frags * 32 lanes * 16 bf16

        pack_B_kernel<<<1, 256, 0, stream>>>(B.dgW, packB);
        {
            const int waves  = (B.Eb + 15) / 16;
            const int blocks = (waves + 3) / 4;  // 4 waves (128 threads) per block
            edge_gemm_wmma<<<blocks, 128, 0, stream>>>(edge_z, B.mask, packB, B.dgb,
                                                       ea, rowmean, B.Eb);
        }
        mean_gather<<<(B.Eb + 255) / 256, 256, 0, stream>>>(rowmean, B.sel, m, B.Eb);
        fill_kernel<<<(B.n + 255) / 256, 256, 0, stream>>>(deg, 1.0f, B.n);
        deg_accum<<<(B.Eb + 255) / 256, 256, 0, stream>>>(B.adjp, B.adjp + B.Eb, m, deg, B.Eb);
        dinv_kernel<<<(B.n + 255) / 256, 256, 0, stream>>>(deg, B.n);
        feat_gemm<<<B.n, B.outDim, 0, stream>>>(node, B.idx, B.w, B.gW, B.outDim, h, B.n);
        gcn_agg<<<B.n, B.outDim, 0, stream>>>(h, m, deg, B.gb, xh, B.n, B.outDim);
        finalize<<<(B.Eb + 255) / 256, 256, 0, stream>>>(ea, xh, B.outDim, B.adjp,
                                                         B.adjp + B.Eb, B.mask,
                                                         B.cW, B.cb, out, B.colBase, B.Eb);
    }
}
